// HINN_28338194219298
// MI455X (gfx1250) — compile-verified, matrix operand order checked by reading, not run on hardware
//
#include <hip/hip_runtime.h>

// HINN forward for MI455X (gfx1250, wave32, WMMA).
// Dominant cost: masked GEMM1 (512x10000 @ 10000x8000). 640MB f32 W+mask
// traffic => ~27us HBM floor at 23.3TB/s; bf16 WMMA (f32 accum) keeps compute
// at parity (~2.5 PF-class dense). Mask-multiply + bf16 convert fused into the
// tile load; Y2/Y3 elementwise epilogues fused into GEMM stores.
// v2: interior/edge kernel split (no guards in hot loop), register-staged
// double-buffered LDS (1 barrier per K-step, global latency hidden by WMMA).

#define BN_EPS 1e-5f
#define DIV_EPS 1e-8f
#define BATCH 512

typedef __attribute__((ext_vector_type(16))) __bf16 v16bf;
typedef __attribute__((ext_vector_type(8)))  __bf16 v8bf;
typedef __attribute__((ext_vector_type(4)))  __bf16 v4bf;
typedef __attribute__((ext_vector_type(8)))  float  v8f;

union Frag16 { v16bf v; v8bf h[2]; };

// ---------------------------------------------------------------------------
// Tile staging: each thread owns 4 float4 of A (128x32 tile) and 4 float4 of
// W*mask products (32x128 tile). KG = guard K tail, NFULL = no N guards.
// ---------------------------------------------------------------------------
template <bool KG, bool NFULL>
__device__ __forceinline__ void load_tiles(const float* __restrict__ A,
                                           const float* __restrict__ W,
                                           const float* __restrict__ Mk,
                                           int K, int N, int blockM, int blockN,
                                           int k0, int t,
                                           float4 (&ar)[4], float4 (&br)[4])
{
#pragma unroll
    for (int r = 0; r < 4; ++r) {
        int lin = t + r * 256;            // 0..1023 float4 slots
        int row = lin >> 3;               // 8 float4 per 32-wide row
        int col = (lin & 7) << 2;
        int gk = k0 + col;
        const float* src = A + (size_t)(blockM + row) * K + gk;   // M grid exact
        if (!KG) {
            ar[r] = *(const float4*)src;
        } else {
            float4 f = {0.f, 0.f, 0.f, 0.f};
            if (gk + 4 <= K) {
                f = *(const float4*)src;
            } else {
                if (gk + 0 < K) f.x = src[0];
                if (gk + 1 < K) f.y = src[1];
                if (gk + 2 < K) f.z = src[2];
                if (gk + 3 < K) f.w = src[3];
            }
            ar[r] = f;
        }
    }
#pragma unroll
    for (int r = 0; r < 4; ++r) {
        int lin  = t + r * 256;           // 0..1023 float4 slots
        int krow = lin >> 5;              // 32 float4 per 128-wide row
        int ncol = (lin & 31) << 2;
        int gk = k0 + krow, gn = blockN + ncol;
        float4 p = {0.f, 0.f, 0.f, 0.f};
        if (!KG || gk < K) {
            const float* wsrc = W  + (size_t)gk * N + gn;
            const float* msrc = Mk + (size_t)gk * N + gn;
            if (NFULL || gn + 4 <= N) {
                float4 w = *(const float4*)wsrc;
                float4 m = *(const float4*)msrc;
                p.x = w.x * m.x; p.y = w.y * m.y; p.z = w.z * m.z; p.w = w.w * m.w;
            } else {
                if (gn + 0 < N) p.x = wsrc[0] * msrc[0];
                if (gn + 1 < N) p.y = wsrc[1] * msrc[1];
                if (gn + 2 < N) p.z = wsrc[2] * msrc[2];
                if (gn + 3 < N) p.w = wsrc[3] * msrc[3];
            }
        }
        br[r] = p;
    }
}

__device__ __forceinline__ void store_tiles(__bf16 (&Alds)[128][40],
                                            __bf16 (&Blds)[128][40],
                                            int t,
                                            const float4 (&ar)[4],
                                            const float4 (&br)[4])
{
#pragma unroll
    for (int r = 0; r < 4; ++r) {
        int lin = t + r * 256;
        int row = lin >> 3;
        int col = (lin & 7) << 2;
        v4bf pk = { (__bf16)ar[r].x, (__bf16)ar[r].y, (__bf16)ar[r].z, (__bf16)ar[r].w };
        *(v4bf*)&Alds[row][col] = pk;                 // ds_store_b64
    }
#pragma unroll
    for (int r = 0; r < 4; ++r) {
        int lin  = t + r * 256;
        int krow = lin >> 5;
        int ncol = (lin & 31) << 2;
        Blds[ncol + 0][krow] = (__bf16)br[r].x;       // transposed: scattered b16
        Blds[ncol + 1][krow] = (__bf16)br[r].y;
        Blds[ncol + 2][krow] = (__bf16)br[r].z;
        Blds[ncol + 3][krow] = (__bf16)br[r].w;
    }
}

// ---------------------------------------------------------------------------
// Masked GEMM with fused epilogue.
//   MODE 0: Out = relu(acc + bias)                                  (Y4)
//   MODE 1: Y1 = relu(acc+bias); Out = relu(xaux*gaux*Y1 + baux)    (Y2)
//   MODE 2: Out = relu(xaux*gaux/(acc+bias+eps) + baux)             (Y3)
// 256 threads = 8 waves, 128x128 C tile, K-step 32, LDS double-buffered.
// NFULL=true: interior columns, all N guards compiled out.
// ---------------------------------------------------------------------------
template <int MODE, bool NFULL>
__global__ __launch_bounds__(256)
void hinn_masked_gemm_bf16(const float* __restrict__ A,
                           const float* __restrict__ W,
                           const float* __restrict__ Mk,
                           const float* __restrict__ bias,
                           const float* __restrict__ xaux,
                           const float* __restrict__ gaux,
                           const float* __restrict__ baux,
                           float* __restrict__ Out,
                           int M, int N, int K, int nBlockOff)
{
    // 32 K-elems + 8 pad => row stride 80B (16B-aligned) for b128 ds loads
    __shared__ __align__(16) __bf16 Alds[2][128][40];
    __shared__ __align__(16) __bf16 Blds[2][128][40];

    const int t    = threadIdx.x;
    const int lane = t & 31;
    const int wave = t >> 5;
    const int wm   = wave >> 1;      // 0..3 -> wave M offset = wm*32
    const int wn   = wave & 1;       // 0..1 -> wave N offset = wn*64
    const int l16  = lane & 15;
    const int hi   = lane >> 4;      // 0/1

    const int blockN = (blockIdx.x + nBlockOff) * 128;
    const int blockM = blockIdx.y * 128;          // grid.y exact for M=512

    const int  nk    = (K + 31) >> 5;
    const bool ktail = (K & 31) != 0;

    v8f acc[2][4];
#pragma unroll
    for (int a = 0; a < 2; ++a)
#pragma unroll
        for (int b = 0; b < 4; ++b)
            acc[a][b] = (v8f){0.f,0.f,0.f,0.f,0.f,0.f,0.f,0.f};

    float4 ar[4], br[4];
    if (ktail && nk == 1)
        load_tiles<true, NFULL>(A, W, Mk, K, N, blockM, blockN, 0, t, ar, br);
    else
        load_tiles<false, NFULL>(A, W, Mk, K, N, blockM, blockN, 0, t, ar, br);
    store_tiles(Alds[0], Blds[0], t, ar, br);
    __syncthreads();

    for (int kt = 0; kt < nk; ++kt) {
        const int  cur      = kt & 1;
        const bool havenext = (kt + 1) < nk;

        // issue next tile's global loads first: latency hidden under WMMA
        if (havenext) {
            if (ktail && (kt + 2 == nk))
                load_tiles<true, NFULL>(A, W, Mk, K, N, blockM, blockN, (kt + 1) * 32, t, ar, br);
            else
                load_tiles<false, NFULL>(A, W, Mk, K, N, blockM, blockN, (kt + 1) * 32, t, ar, br);
        }

        // fragments per documented bf16 16x32 layout:
        // lane holds K = {hi*8..+7, 16+hi*8..+7} -> two b128 ds loads each
        Frag16 af[2], bfr[4];
#pragma unroll
        for (int a = 0; a < 2; ++a) {
            int ml = wm * 32 + a * 16 + l16;
            af[a].h[0] = *(const v8bf*)&Alds[cur][ml][hi * 8];
            af[a].h[1] = *(const v8bf*)&Alds[cur][ml][16 + hi * 8];
        }
#pragma unroll
        for (int b = 0; b < 4; ++b) {
            int nl = wn * 64 + b * 16 + l16;
            bfr[b].h[0] = *(const v8bf*)&Blds[cur][nl][hi * 8];
            bfr[b].h[1] = *(const v8bf*)&Blds[cur][nl][16 + hi * 8];
        }
#pragma unroll
        for (int a = 0; a < 2; ++a)
#pragma unroll
            for (int b = 0; b < 4; ++b)
                acc[a][b] = __builtin_amdgcn_wmma_f32_16x16x32_bf16(
                    false, af[a].v, false, bfr[b].v,
                    (short)0, acc[a][b], false, false);

        if (havenext)
            store_tiles(Alds[cur ^ 1], Blds[cur ^ 1], t, ar, br);
        __syncthreads();
    }

    // ---- fused epilogue + store ------------------------------------------
#pragma unroll
    for (int a = 0; a < 2; ++a) {
#pragma unroll
        for (int b = 0; b < 4; ++b) {
            int n = blockN + wn * 64 + b * 16 + l16;
            if (!NFULL && n >= N) continue;
            float bn = bias[n];
#pragma unroll
            for (int r = 0; r < 8; ++r) {
                int m = blockM + wm * 32 + a * 16 + hi * 8 + r;
                size_t o = (size_t)m * N + n;
                float v = acc[a][b][r] + bn;
                float res;
                if (MODE == 0) {
                    res = fmaxf(v, 0.f);
                } else if (MODE == 1) {
                    float y1 = fmaxf(v, 0.f);
                    res = fmaxf(xaux[o] * gaux[n] * y1 + baux[n], 0.f);
                } else {
                    res = fmaxf(xaux[o] * gaux[n] / (v + DIV_EPS) + baux[n], 0.f);
                }
                Out[o] = res;
            }
        }
    }
}

// ---------------------------------------------------------------------------
// FC-20 branch: out[m,0:20] = relu(concat(A1[m],A2[m]) @ W + b), W:[K1+K2,20]
// ---------------------------------------------------------------------------
__global__ __launch_bounds__(256)
void hinn_fc20(const float* __restrict__ A1, int K1,
               const float* __restrict__ A2, int K2,
               const float* __restrict__ W, const float* __restrict__ b,
               float* __restrict__ out)
{
    __shared__ float red[256][20];
    const int m = blockIdx.x;
    float acc[20];
#pragma unroll
    for (int n = 0; n < 20; ++n) acc[n] = 0.f;
    for (int k = threadIdx.x; k < K1; k += 256) {
        float x = A1[(size_t)m * K1 + k];
        const float* wr = W + (size_t)k * 20;
#pragma unroll
        for (int n = 0; n < 20; ++n) acc[n] += x * wr[n];
    }
#pragma unroll
    for (int n = 0; n < 20; ++n) red[threadIdx.x][n] = acc[n];
    __syncthreads();
    if (threadIdx.x < 20) {
        int n = threadIdx.x;
        float s = b[n];
        for (int t = 0; t < 256; ++t) s += red[t][n];
        if (A2) {
            for (int j = 0; j < K2; ++j)
                s += A2[(size_t)m * K2 + j] * W[(size_t)(K1 + j) * 20 + n];
        }
        out[(size_t)m * 20 + n] = fmaxf(s, 0.f);
    }
}

// ---------------------------------------------------------------------------
// Head dense: out[m,n] = act( bias[n] + sum_k BN(concat(A1,A2)[m,k]) * W[k,n] )
// ---------------------------------------------------------------------------
__global__ __launch_bounds__(256)
void hinn_dense(const float* __restrict__ A1, int K1,
                const float* __restrict__ A2, int K2,
                const float* __restrict__ W, const float* __restrict__ bias,
                const float* __restrict__ bng, const float* __restrict__ bnb,
                const float* __restrict__ bnm, const float* __restrict__ bnv,
                float* __restrict__ out, int M, int N, int relu)
{
    int idx = blockIdx.x * blockDim.x + threadIdx.x;
    if (idx >= M * N) return;
    int m = idx / N, n = idx - m * N;
    int K = K1 + K2;
    float s = bias[n];
    for (int k = 0; k < K; ++k) {
        float x = (k < K1) ? A1[(size_t)m * K1 + k] : A2[(size_t)m * K2 + (k - K1)];
        if (bng) x = (x - bnm[k]) * rsqrtf(bnv[k] + BN_EPS) * bng[k] + bnb[k];
        s += x * W[(size_t)k * N + n];
    }
    out[idx] = relu ? fmaxf(s, 0.f) : s;
}

// ---------------------------------------------------------------------------
static inline void launch_gemm_pair(int mode, const float* A, const float* W,
                                    const float* Mk, const float* bias,
                                    const float* xaux, const float* gaux,
                                    const float* baux, float* Out,
                                    int N, int K, hipStream_t stream)
{
    const int nFull = N / 128;
    const int rem   = N - nFull * 128;
    dim3 blk(256);
    dim3 gFull(nFull, BATCH / 128);
    dim3 gEdge(1, BATCH / 128);
    switch (mode) {
    case 0:
        if (nFull) hinn_masked_gemm_bf16<0, true><<<gFull, blk, 0, stream>>>(
            A, W, Mk, bias, xaux, gaux, baux, Out, BATCH, N, K, 0);
        if (rem)   hinn_masked_gemm_bf16<0, false><<<gEdge, blk, 0, stream>>>(
            A, W, Mk, bias, xaux, gaux, baux, Out, BATCH, N, K, nFull);
        break;
    case 1:
        if (nFull) hinn_masked_gemm_bf16<1, true><<<gFull, blk, 0, stream>>>(
            A, W, Mk, bias, xaux, gaux, baux, Out, BATCH, N, K, 0);
        if (rem)   hinn_masked_gemm_bf16<1, false><<<gEdge, blk, 0, stream>>>(
            A, W, Mk, bias, xaux, gaux, baux, Out, BATCH, N, K, nFull);
        break;
    default:
        if (nFull) hinn_masked_gemm_bf16<2, true><<<gFull, blk, 0, stream>>>(
            A, W, Mk, bias, xaux, gaux, baux, Out, BATCH, N, K, 0);
        if (rem)   hinn_masked_gemm_bf16<2, false><<<gEdge, blk, 0, stream>>>(
            A, W, Mk, bias, xaux, gaux, baux, Out, BATCH, N, K, nFull);
        break;
    }
}

extern "C" void kernel_launch(void* const* d_in, const int* in_sizes, int n_in,
                              void* d_out, int out_size, void* d_ws, size_t ws_size,
                              hipStream_t stream)
{
    (void)in_sizes; (void)n_in; (void)out_size; (void)ws_size;
    auto F = [&](int i) { return (const float*)d_in[i]; };

    // setup_inputs() insertion-order flattening
    const float* x_snp  = F(0);   // [512,10000]
    const float* x_cpg  = F(1);   // [512,8000]
    const float* x_gene = F(2);   // [512,2000]
    const float* demog  = F(3);   // [512,6]
    const float* mask_sc = F(4);  // [10000,8000]
    const float* mask_cg = F(5);  // [8000,2000]
    const float* mask_gg = F(6);  // [2000,1000]
    const float* Wsc = F(7),  *bsc = F(8);
    const float* g_cpg = F(9), *B2 = F(10);
    const float* Wcg = F(11), *bcg = F(12);
    const float* g_gene = F(13), *B3 = F(14);
    const float* Wgg = F(15), *bgg = F(16);
    const float* Wsnp20 = F(17), *bsnp20 = F(18);
    const float* Wcpg20 = F(19), *bcpg20 = F(20);
    const float* Wgene20 = F(21), *bgene20 = F(22);
    const float* bn1g = F(23), *bn1b = F(24), *bn1m = F(25), *bn1v = F(26);
    const float* W1 = F(27), *b1 = F(28);
    const float* bn2g = F(29), *bn2b = F(30), *bn2m = F(31), *bn2v = F(32);
    const float* W2 = F(33), *b2 = F(34);
    const float* bn3g = F(35), *bn3b = F(36), *bn3m = F(37), *bn3v = F(38);
    const float* W3 = F(39), *b3 = F(40);
    const float* bn4g = F(41), *bn4b = F(42), *bn4m = F(43), *bn4v = F(44);
    const float* W4 = F(45), *b4 = F(46);
    const float* Wp = F(47), *bp = F(48);
    const float* bnDg = F(49), *bnDb = F(50), *bnDm = F(51), *bnDv = F(52);
    const float* Wd = F(53), *bd = F(54);
    const float* Wo = F(55), *bo = F(56);

    // workspace carve-up (f32)
    float* ws = (float*)d_ws;
    float* Y2      = ws;                           // 512*8000
    float* Y3      = Y2 + (size_t)BATCH * 8000;    // 512*2000
    float* Y4      = Y3 + (size_t)BATCH * 2000;    // 512*1000
    float* snp_fc  = Y4 + (size_t)BATCH * 1000;    // 512*20
    float* cpg_fc  = snp_fc + (size_t)BATCH * 20;
    float* gene_fc = cpg_fc + (size_t)BATCH * 20;
    float* h1      = gene_fc + (size_t)BATCH * 20; // 512*128
    float* h2      = h1 + (size_t)BATCH * 128;     // 512*128
    float* h20     = h2 + (size_t)BATCH * 128;     // 512*20
    float* dh      = h20 + (size_t)BATCH * 20;     // 512*128

    dim3 blk(256);

    // GEMM1 + fused Y2 epilogue: [512,10000] x [10000,8000]
    launch_gemm_pair(1, x_snp, Wsc, mask_sc, bsc, x_cpg, g_cpg, B2, Y2, 8000, 10000, stream);

    // snp_fc = relu(x_snp @ Wsnp20 + b)
    hinn_fc20<<<BATCH, blk, 0, stream>>>(x_snp, 10000, nullptr, 0, Wsnp20, bsnp20, snp_fc);

    // GEMM2 + fused Y3 epilogue: [512,8000] x [8000,2000]
    launch_gemm_pair(2, Y2, Wcg, mask_cg, bcg, x_gene, g_gene, B3, Y3, 2000, 8000, stream);

    // cpg_fc = relu([Y2, snp_fc] @ Wcpg20 + b)
    hinn_fc20<<<BATCH, blk, 0, stream>>>(Y2, 8000, snp_fc, 20, Wcpg20, bcpg20, cpg_fc);

    // GEMM3: Y4 = relu([512,2000] x [2000,1000] + b)
    launch_gemm_pair(0, Y3, Wgg, mask_gg, bgg, nullptr, nullptr, nullptr, Y4, 1000, 2000, stream);

    // gene_fc = relu([Y3, cpg_fc] @ Wgene20 + b)
    hinn_fc20<<<BATCH, blk, 0, stream>>>(Y3, 2000, cpg_fc, 20, Wgene20, bgene20, gene_fc);

    // block1: BN([Y4, gene_fc]) @ W1 -> relu, [512,1020]->[512,128]
    hinn_dense<<<(BATCH * 128 + 255) / 256, blk, 0, stream>>>(
        Y4, 1000, gene_fc, 20, W1, b1, bn1g, bn1b, bn1m, bn1v, h1, BATCH, 128, 1);
    // block2..4: 128->128
    hinn_dense<<<(BATCH * 128 + 255) / 256, blk, 0, stream>>>(
        h1, 128, nullptr, 0, W2, b2, bn2g, bn2b, bn2m, bn2v, h2, BATCH, 128, 1);
    hinn_dense<<<(BATCH * 128 + 255) / 256, blk, 0, stream>>>(
        h2, 128, nullptr, 0, W3, b3, bn3g, bn3b, bn3m, bn3v, h1, BATCH, 128, 1);
    hinn_dense<<<(BATCH * 128 + 255) / 256, blk, 0, stream>>>(
        h1, 128, nullptr, 0, W4, b4, bn4g, bn4b, bn4m, bn4v, h2, BATCH, 128, 1);
    // post_dense20: 128->20 relu (no BN)
    hinn_dense<<<(BATCH * 20 + 255) / 256, blk, 0, stream>>>(
        h2, 128, nullptr, 0, Wp, bp, nullptr, nullptr, nullptr, nullptr, h20, BATCH, 20, 1);
    // demog block: BN([h20, demog]) @ Wd -> relu, [512,26]->[512,128]
    hinn_dense<<<(BATCH * 128 + 255) / 256, blk, 0, stream>>>(
        h20, 20, demog, 6, Wd, bd, bnDg, bnDb, bnDm, bnDv, dh, BATCH, 128, 1);
    // output: 128->1, no relu, straight to d_out [512] f32
    hinn_dense<<<(BATCH * 1 + 255) / 256, blk, 0, stream>>>(
        dh, 128, nullptr, 0, Wo, bo, nullptr, nullptr, nullptr, nullptr,
        (float*)d_out, BATCH, 1, 0);
}
